// EdgeFeatureGATv2Conv_86543591014919
// MI455X (gfx1250) — matrix operand order repaired and checked by reading, not compiled, not run on hardware
//
#include <hip/hip_runtime.h>

// ---------------- types / constants ----------------
typedef __attribute__((ext_vector_type(16))) __bf16 v16bf;
typedef __attribute__((ext_vector_type(8)))  float  v8f;

#define IN_CH 256   // input channels
#define HC    256   // H*C
#define HH    4     // heads
#define EDIM  32    // edge_dim
#define NEG_SLOPE 0.2f

static __device__ __forceinline__ unsigned short f2bf(float f) {
  union { float f; unsigned u; } v; v.f = f;
  unsigned u = v.u;
  u += 0x7FFFu + ((u >> 16) & 1u);      // round-to-nearest-even
  return (unsigned short)(u >> 16);
}

// bit-compatible view of a v16bf operand (all trivial members -> legal union)
union ABf16 {
  v16bf v;
  unsigned short h[16];
  unsigned int   w[8];
};

// A-operand K order this code uses (contiguous k = 16*half + e, same as B swizzle; not AMD's table order): element e of lane -> K offset within 32
static __device__ __forceinline__ int a_kmap(int e, int half) {
  return e + (half ? 16 : 0);
}

// ---------------- pre-swizzle A operand (row-major f32 -> bf16 WMMA A-lane layout) ----------------
// As[((rt*ksteps + ks)*32 + lane)*16 + e] = bf16(X[(rt*16 + (lane&15)) * K + ks*32 + a_kmap(e,lane>>4)])
__global__ void k_swizzleA(const float* __restrict__ X, unsigned short* __restrict__ As,
                           int R, int K) {
  int tid = blockIdx.x * blockDim.x + threadIdx.x;
  int ksteps = K >> 5;
  int rtiles = (R + 15) >> 4;
  int total = rtiles * ksteps * 512;
  if (tid >= total) return;
  int e    = tid & 15;
  int lane = (tid >> 4) & 31;
  int rest = tid >> 9;               // rt*ksteps + ks
  int ks = rest % ksteps;
  int rt = rest / ksteps;
  int row = rt * 16 + (lane & 15); if (row >= R) row = R - 1;
  int k   = ks * 32 + a_kmap(e, lane >> 4);
  As[tid] = f2bf(X[(size_t)row * K + k]);
}

// ---------------- pre-swizzle B operand (weights) ----------------
// Ws[((nt*ksteps + ks)*32 + lane)*16 + e] = bf16(W[ks*32 + e + 16*(lane>=16)][nt*16 + (lane&15)])
__global__ void k_swizzleB(const float* __restrict__ W, unsigned short* __restrict__ Ws,
                           int K, int Ncols) {
  int tid = blockIdx.x * blockDim.x + threadIdx.x;
  int total = K * Ncols;
  if (tid >= total) return;
  int e    = tid & 15;
  int lane = (tid >> 4) & 31;
  int rest = tid >> 9;               // nt*ksteps + ks
  int ksteps = K >> 5;
  int ks = rest % ksteps;
  int nt = rest / ksteps;
  int k  = ks * 32 + e + ((lane >> 4) << 4);
  int n  = nt * 16 + (lane & 15);
  Ws[tid] = f2bf(W[(size_t)k * Ncols + n]);
}

// ---------------- init: out = bias, amax = -inf, denom = 0 ----------------
__global__ void k_init(float* __restrict__ out, const float* __restrict__ bias,
                       float* __restrict__ amax, float* __restrict__ denom, int N) {
  int tid = blockIdx.x * blockDim.x + threadIdx.x;
  if (tid < N * HC) out[tid] = bias[tid & (HC - 1)];
  if (tid < N * HH) { amax[tid] = -INFINITY; denom[tid] = 0.f; }
}

// ---------------- node GEMM: out[N,256] = x @ W + b ; 16x64 tile per wave ----------------
__global__ __launch_bounds__(32)
void k_node_gemm(const unsigned short* __restrict__ Xs, const unsigned short* __restrict__ Ws,
                 const float* __restrict__ bvec, float* __restrict__ out, int N) {
  const int lane = threadIdx.x;
  const int half = lane >> 4;        // 0/1
  const int l15  = lane & 15;
  const int mt   = blockIdx.x;       // row tile
  const int m0   = mt * 16;
  const int nt0  = blockIdx.y * 4;   // 4 column tiles per wave

  v8f c[4];
  #pragma unroll
  for (int j = 0; j < 4; ++j) {
    float bv = bvec[(nt0 + j) * 16 + l15];
    #pragma unroll
    for (int r = 0; r < 8; ++r) c[j][r] = bv;
  }

  #pragma unroll
  for (int ks = 0; ks < 8; ++ks) {
    ABf16 a;
    const unsigned int* ap = (const unsigned int*)(Xs + ((size_t)(mt * 8 + ks) * 32 + lane) * 16);
    #pragma unroll
    for (int i = 0; i < 8; ++i) a.w[i] = ap[i];   // 32 contiguous bytes -> 2x b128
    #pragma unroll
    for (int j = 0; j < 4; ++j) {
      ABf16 b;
      const unsigned int* bp =
          (const unsigned int*)(Ws + ((size_t)((nt0 + j) * 8 + ks) * 32 + lane) * 16);
      #pragma unroll
      for (int i = 0; i < 8; ++i) b.w[i] = bp[i];
      c[j] = __builtin_amdgcn_wmma_f32_16x16x32_bf16(false, a.v, false, b.v, (short)0, c[j],
                                                     false, false);
    }
  }
  #pragma unroll
  for (int j = 0; j < 4; ++j) {
    #pragma unroll
    for (int r = 0; r < 8; ++r) {
      int row = m0 + r + half * 8;
      if (row < N) out[(size_t)row * HC + (nt0 + j) * 16 + l15] = c[j][r];
    }
  }
}

// ---------------- fused edge kernel ----------------
// ee = relu(edge_attr @ Wee + b_ee)   (WMMA, -> LDS bf16 16x128)
// e  = ee @ Wle                        (WMMA, registers per col-tile)
// m  = leakyrelu(e + x_l[src] + x_r[dst]) ; alpha[e,h] = sum_c m*att
__global__ __launch_bounds__(32)
void k_edge_fused(const unsigned short* __restrict__ EAs, const unsigned short* __restrict__ Wee_s,
                  const float* __restrict__ b_ee, const unsigned short* __restrict__ Wle_s,
                  const float* __restrict__ x_l, const float* __restrict__ x_r,
                  const float* __restrict__ att, const int* __restrict__ ei,
                  float* __restrict__ alpha, int E) {
  __shared__ unsigned short ee_lds[16 * 128];
  __shared__ int src_sh[16];
  __shared__ int dst_sh[16];

  const int lane = threadIdx.x;
  const int half = lane >> 4;
  const int l15  = lane & 15;
  const int tile = blockIdx.x;
  const int e0   = tile * 16;

  if (lane < 16) {
    int e_idx = e0 + lane; if (e_idx >= E) e_idx = E - 1;
    src_sh[lane] = ei[e_idx];
    dst_sh[lane] = ei[E + e_idx];
  }

  // A operand from pre-swizzled edge_attr (K=32, one step per tile)
  ABf16 a_ea;
  {
    const unsigned int* ap = (const unsigned int*)(EAs + ((size_t)tile * 32 + lane) * 16);
    #pragma unroll
    for (int i = 0; i < 8; ++i) a_ea.w[i] = ap[i];
  }

  // Stage 1: ee tile (16 edges x 128) -> LDS (bf16)
  #pragma unroll
  for (int nt = 0; nt < 8; ++nt) {
    v8f c;
    float bv = b_ee[nt * 16 + l15];
    #pragma unroll
    for (int r = 0; r < 8; ++r) c[r] = bv;
    ABf16 b;
    const unsigned int* bp = (const unsigned int*)(Wee_s + ((size_t)nt * 32 + lane) * 16);
    #pragma unroll
    for (int i = 0; i < 8; ++i) b.w[i] = bp[i];
    c = __builtin_amdgcn_wmma_f32_16x16x32_bf16(false, a_ea.v, false, b.v, (short)0, c, false, false);
    int col = nt * 16 + l15;
    #pragma unroll
    for (int r = 0; r < 8; ++r) {
      int row = r + half * 8;
      float v = c[r] > 0.f ? c[r] : 0.f;      // ReLU
      ee_lds[row * 128 + col] = f2bf(v);
    }
  }
  __syncthreads();

  // Stage 2: per head, 4 col-tiles of lin_edge; fuse message + att reduction
  #pragma unroll 1
  for (int h = 0; h < HH; ++h) {
    float acc[8];
    #pragma unroll
    for (int r = 0; r < 8; ++r) acc[r] = 0.f;

    #pragma unroll 1
    for (int ntl = 0; ntl < 4; ++ntl) {
      int nt  = h * 4 + ntl;
      int col = nt * 16 + l15;
      v8f c;
      #pragma unroll
      for (int r = 0; r < 8; ++r) c[r] = 0.f;
      #pragma unroll
      for (int ks = 0; ks < 4; ++ks) {
        ABf16 a, b;
        #pragma unroll
        for (int p = 0; p < 8; ++p) {
          int k = ks * 32 + a_kmap(2 * p, half);
          a.w[p] = *(const unsigned int*)(&ee_lds[l15 * 128 + k]);
        }
        const unsigned int* bp = (const unsigned int*)(Wle_s + ((size_t)(nt * 4 + ks) * 32 + lane) * 16);
        #pragma unroll
        for (int i = 0; i < 8; ++i) b.w[i] = bp[i];
        c = __builtin_amdgcn_wmma_f32_16x16x32_bf16(false, a.v, false, b.v, (short)0, c, false, false);
      }
      float attv = att[col];                  // att flat index == column index
      #pragma unroll
      for (int r = 0; r < 8; ++r) {
        int row = r + half * 8;
        float m = c[r] + x_l[(size_t)src_sh[row] * HC + col]
                       + x_r[(size_t)dst_sh[row] * HC + col];
        m = m > 0.f ? m : NEG_SLOPE * m;      // leaky relu
        acc[r] += m * attv;
      }
    }
    // reduce across the 16 lanes sharing each edge row (xor masks stay inside half-wave)
    #pragma unroll
    for (int r = 0; r < 8; ++r) {
      float v = acc[r];
      v += __shfl_xor(v, 1, 32);
      v += __shfl_xor(v, 2, 32);
      v += __shfl_xor(v, 4, 32);
      v += __shfl_xor(v, 8, 32);
      if (l15 == 0) {
        int e_idx = e0 + r + half * 8;
        if (e_idx < E) alpha[(size_t)e_idx * HH + h] = v;
      }
    }
  }
}

// ---------------- segment softmax + scatter ----------------
__global__ void k_segmax(const float* __restrict__ alpha, const int* __restrict__ ei,
                         float* __restrict__ amax, int E) {
  int tid = blockIdx.x * blockDim.x + threadIdx.x;
  if (tid >= E * HH) return;
  int e = tid >> 2, h = tid & 3;
  int dst = ei[E + e];
  atomicMax(&amax[dst * HH + h], alpha[tid]);
}

__global__ void k_expsum(float* __restrict__ alpha, const int* __restrict__ ei,
                         const float* __restrict__ amax, float* __restrict__ denom, int E) {
  int tid = blockIdx.x * blockDim.x + threadIdx.x;
  if (tid >= E * HH) return;
  int e = tid >> 2, h = tid & 3;
  int dst = ei[E + e];
  float ex = expf(alpha[tid] - amax[dst * HH + h]);
  alpha[tid] = ex;                          // overwrite with exp
  atomicAdd(&denom[dst * HH + h], ex);
}

__global__ __launch_bounds__(256)
void k_scatter(const float* __restrict__ alpha, const float* __restrict__ denom,
               const int* __restrict__ ei, const float* __restrict__ x_l,
               float* __restrict__ out, int E) {
  int e = blockIdx.x;
  int c = threadIdx.x;
  int src = ei[e], dst = ei[E + e];
  int h = c >> 6;
  float a = alpha[(size_t)e * HH + h] / (denom[dst * HH + h] + 1e-16f);
  atomicAdd(&out[(size_t)dst * HC + c], x_l[(size_t)src * HC + c] * a);
}

// ---------------- launcher ----------------
extern "C" void kernel_launch(void* const* d_in, const int* in_sizes, int n_in,
                              void* d_out, int out_size, void* d_ws, size_t ws_size,
                              hipStream_t stream) {
  const float* x    = (const float*)d_in[0];
  const float* ea   = (const float*)d_in[1];
  const float* W_l  = (const float*)d_in[2];
  const float* b_l  = (const float*)d_in[3];
  const float* W_r  = (const float*)d_in[4];
  const float* b_r  = (const float*)d_in[5];
  const float* W_ee = (const float*)d_in[6];
  const float* b_ee = (const float*)d_in[7];
  const float* W_le = (const float*)d_in[8];
  const float* att  = (const float*)d_in[9];
  const float* bias = (const float*)d_in[10];
  const int*   ei   = (const int*)d_in[11];
  float* out = (float*)d_out;

  const int N = in_sizes[0] / IN_CH;
  const int E = in_sizes[1] / EDIM;
  const int NT = (N + 15) / 16;
  const int ET = (E + 15) / 16;

  char* ws = (char*)d_ws;
  size_t off = 0;
  auto alloc = [&](size_t bytes) -> void* {
    void* p = ws + off;
    off = (off + bytes + 255) & ~(size_t)255;
    return p;
  };
  unsigned short* Wl_s  = (unsigned short*)alloc((size_t)IN_CH * HC * 2);
  unsigned short* Wr_s  = (unsigned short*)alloc((size_t)IN_CH * HC * 2);
  unsigned short* Wee_s = (unsigned short*)alloc((size_t)EDIM * (EDIM * HH) * 2);
  unsigned short* Wle_s = (unsigned short*)alloc((size_t)(EDIM * HH) * HC * 2);
  unsigned short* Xs    = (unsigned short*)alloc((size_t)NT * 16 * IN_CH * 2);
  unsigned short* EAs   = (unsigned short*)alloc((size_t)ET * 16 * EDIM * 2);
  float* x_l_buf = (float*)alloc((size_t)N * HC * 4);
  float* x_r_buf = (float*)alloc((size_t)N * HC * 4);
  float* alpha   = (float*)alloc((size_t)E * HH * 4);
  float* amax    = (float*)alloc((size_t)N * HH * 4);
  float* denom   = (float*)alloc((size_t)N * HH * 4);
  (void)ws_size; (void)n_in; (void)out_size;

  // 1) pre-swizzle weights (B operands) and activations (A operands) into WMMA lane layouts
  k_swizzleB<<<(IN_CH * HC + 255) / 256, 256, 0, stream>>>(W_l, Wl_s, IN_CH, HC);
  k_swizzleB<<<(IN_CH * HC + 255) / 256, 256, 0, stream>>>(W_r, Wr_s, IN_CH, HC);
  k_swizzleB<<<(EDIM * EDIM * HH + 255) / 256, 256, 0, stream>>>(W_ee, Wee_s, EDIM, EDIM * HH);
  k_swizzleB<<<(EDIM * HH * HC + 255) / 256, 256, 0, stream>>>(W_le, Wle_s, EDIM * HH, HC);
  k_swizzleA<<<((size_t)NT * 16 * IN_CH + 255) / 256, 256, 0, stream>>>(x, Xs, N, IN_CH);
  k_swizzleA<<<((size_t)ET * 16 * EDIM + 255) / 256, 256, 0, stream>>>(ea, EAs, E, EDIM);

  // 2) init out=bias, amax=-inf, denom=0
  k_init<<<((size_t)N * HC + 255) / 256, 256, 0, stream>>>(out, bias, amax, denom, N);

  // 3) node transforms (WMMA bf16, 16x64 tile per wave)
  k_node_gemm<<<dim3(NT, 4), 32, 0, stream>>>(Xs, Wl_s, b_l, x_l_buf, N);
  k_node_gemm<<<dim3(NT, 4), 32, 0, stream>>>(Xs, Wr_s, b_r, x_r_buf, N);

  // 4) fused edge pipeline -> alpha[E,4] (never materializes e[E,256])
  k_edge_fused<<<ET, 32, 0, stream>>>(EAs, Wee_s, b_ee, Wle_s,
                                      x_l_buf, x_r_buf, att, ei, alpha, E);

  // 5) segment softmax
  k_segmax<<<((size_t)E * HH + 255) / 256, 256, 0, stream>>>(alpha, ei, amax, E);
  k_expsum<<<((size_t)E * HH + 255) / 256, 256, 0, stream>>>(alpha, ei, amax, denom, E);

  // 6) weighted scatter-aggregate onto out (pre-seeded with bias)
  k_scatter<<<E, 256, 0, stream>>>(alpha, denom, ei, x_l_buf, out, E);
}
